// Local_attention_70111046140243
// MI455X (gfx1250) — compile-verified, matrix operand order checked by reading, not run on hardware
//
#include <hip/hip_runtime.h>

// MI455X (gfx1250) Local_attention. Compute-bound (~33 GFLOP vs ~15 MB L2-resident
// traffic) -> feed the WMMA pipes. Both GEMMs use v_wmma_f32_16x16x32_f16 with
// K reordered as (tap, channel) so every K-step is one spatial tap x 32 channels:
// each fragment is two contiguous ds_load_b128, window padding is handled by a
// single address select onto a zero page (no data masking ops).

typedef __attribute__((ext_vector_type(16))) _Float16 v16h;
typedef __attribute__((ext_vector_type(8)))  _Float16 v8h;
typedef __attribute__((ext_vector_type(8)))  float    v8f;

#define IMG    128
#define CH     32
#define TILE   16                  // pixels per workgroup (one row segment)
#define SLABW  22                  // TILE + 7 - 1 columns of x
#define NWAVES 8

// LDS layout (bytes):
//   s_slab : f16[7][22][32]         =   9856   (x window slab, channel-innermost)
//   s_af   : f16[9][2][32][16]      =  18432   (w1 pre-packed in fragment order)
//   s_w3t  : f16[32][49][32]        = 100352   (w3 transposed to [o][tap][ch])
//   s_att  : f32[784]               =   3136
//   s_out  : f32[512]               =   2048   (also the zero page during GEMM1)
//   total 133824 B  (< 320 KB / WGP)
#define OFF_AF   9856
#define OFF_W3T  28288
#define OFF_ATT  128640
#define OFF_OUT  131776
#define SMEM_BYTES 133824

__device__ __forceinline__ v8f wmma_f16(v16h a, v16h b, v8f c) {
    return __builtin_amdgcn_wmma_f32_16x16x32_f16(
        false, a, false, b, (short)0, c, false, false);
}
__device__ __forceinline__ v16h cat8(v8h lo, v8h hi) {
    return __builtin_shufflevector(lo, hi,
        0, 1, 2, 3, 4, 5, 6, 7, 8, 9, 10, 11, 12, 13, 14, 15);
}
__device__ __forceinline__ v16h bcast16(_Float16 x) {
    v16h r;
    #pragma unroll
    for (int i = 0; i < 16; ++i) r[i] = x;
    return r;
}

__global__ __launch_bounds__(256)
void local_attn_kernel(const float* __restrict__ x,
                       const float* __restrict__ w1,
                       const float* __restrict__ b1,
                       const float* __restrict__ w2,
                       const float* __restrict__ b2,
                       const float* __restrict__ w3,
                       const float* __restrict__ b3,
                       float* __restrict__ out,        // [2,32,128,128]
                       float* __restrict__ att_out) {  // [2,1,16384,7,7]
    extern __shared__ char smem[];
    _Float16* s_slab = (_Float16*)smem;                 // [row][col][ch]
    _Float16* s_af   = (_Float16*)(smem + OFF_AF);      // [kk][mt][lane][16]
    _Float16* s_w3t  = (_Float16*)(smem + OFF_W3T);     // [o][tap][ch]
    float*    s_att  = (float*)(smem + OFF_ATT);
    float*    s_out  = (float*)(smem + OFF_OUT);
    const _Float16* s_zero = (const _Float16*)(smem + OFF_OUT); // zero during GEMM1

    const int tid  = threadIdx.x;
    const int lane = tid & 31;
    const int wave = tid >> 5;
    const int lg8  = (lane >> 4) << 3;   // WMMA 16-bit lane-group K offset

    const int bid  = blockIdx.x;         // 2*128*8 = 2048 blocks
    const int bimg = bid >> 10;
    const int rem  = bid & 1023;
    const int h    = rem >> 3;
    const int w0   = (rem & 7) << 4;

    // ---- Stage LDS -------------------------------------------------------
    // x slab, channel-innermost, coalesced-ish reads along image columns
    for (int i = tid; i < CH * 7 * SLABW; i += 256) {
        int cc = i % SLABW;
        int j  = i / SLABW;
        int aa = j % 7, c = j / 7;
        int hh = h + aa - 3, ww = w0 + cc - 3;
        float v = 0.f;
        if ((unsigned)hh < (unsigned)IMG && (unsigned)ww < (unsigned)IMG)
            v = x[(((size_t)bimg * CH + c) * IMG + hh) * IMG + ww];
        s_slab[(aa * SLABW + cc) * CH + c] = (_Float16)v;
    }
    // w1 packed directly in WMMA A-fragment order: frag[e] <-> ch = lg8 + (e<8?e:e+8)
    for (int i = tid; i < 9 * 2 * 32 * 16; i += 256) {
        int e  = i & 15;
        int l  = (i >> 4) & 31;
        int f  = i >> 9;           // kk*2 + mt
        int mt = f & 1, kk = f >> 1;
        int c  = ((e & 8) ? e + 8 : e) + ((l >> 4) << 3);
        int o  = mt * 16 + (l & 15);
        s_af[i] = (_Float16)w1[(o * CH + c) * 9 + kk];
    }
    // w3 transposed [o][tap][ch]; coalesced float4 global reads, LDS scatter
    for (int i = tid; i < (CH * 1568) / 4; i += 256) {
        float4 wv = ((const float4*)w3)[i];
        #pragma unroll
        for (int u = 0; u < 4; ++u) {
            int ff = i * 4 + u;
            int o  = ff / 1568;
            int r  = ff % 1568;
            int c  = r / 49, q = r % 49;
            s_w3t[(o * 49 + q) * CH + c] = (_Float16)(&wv.x)[u];
        }
    }
    for (int i = tid; i < 512; i += 256) s_out[i] = 0.f;  // reduction buf + zero page
    __syncthreads();

    // sigmoid-argument constant: sum_o w2[o]*b1[o] + b2
    float bt = w2[lane] * b1[lane];
    #pragma unroll
    for (int off = 16; off >= 1; off >>= 1) bt += __shfl_xor(bt, off, 32);
    const float bias_att = bt + b2[0];

    float w2lo[8], w2hi[8];
    #pragma unroll
    for (int r = 0; r < 8; ++r) {
        w2lo[r] = w2[r + lg8];
        w2hi[r] = w2[16 + r + lg8];
    }

    // ---- GEMM1 + fused att: k = W1[32x288] * im2col[288x784] -------------
    for (int nt = wave; nt < 49; nt += NWAVES) {
        const int col = nt * 16 + (lane & 15);   // (pixel, window-pos) column
        const int n   = col / 49;
        const int q   = col % 49;
        const int wi  = q / 7, wj = q % 7;

        v8f acc0 = {}, acc1 = {};
        #pragma unroll
        for (int kk = 0; kk < 9; ++kk) {         // one 3x3 tap per K-step
            const int di = kk / 3, dj = kk % 3;  // compile-time after unroll
            int aa = wi + di - 1, bb = wj + dj - 1;
            bool valid = ((unsigned)aa < 7u) && ((unsigned)bb < 7u);
            // window padding: one address select onto the zero page (s_out==0 here)
            const _Float16* bp =
                valid ? (s_slab + ((aa * SLABW + n + bb) * CH + lg8)) : s_zero;
            v16h bf = cat8(*(const v8h*)bp, *(const v8h*)(bp + 16));
            v16h a0 = *(const v16h*)(s_af + (kk * 2 + 0) * 512 + lane * 16);
            v16h a1 = *(const v16h*)(s_af + (kk * 2 + 1) * 512 + lane * 16);
            acc0 = wmma_f16(a0, bf, acc0);
            acc1 = wmma_f16(a1, bf, acc1);
        }
        // att = sigmoid(sum_o w2[o]*k[o,col] + bias); vgpr r holds M=r+8*(lane/16)
        float partial = 0.f;
        #pragma unroll
        for (int r = 0; r < 8; ++r)
            partial += w2lo[r] * acc0[r] + w2hi[r] * acc1[r];
        partial += __shfl_xor(partial, 16, 32);
        float att = 1.f / (1.f + __expf(-(partial + bias_att)));
        if (lane < 16) {
            s_att[col] = att;
            att_out[((size_t)bimg * (IMG * IMG) + (h * IMG + w0 + n)) * 49 + q] = att;
        }
    }
    __syncthreads();

    // ---- GEMM2: out = W3[32x1568] * (lx*(1+att))[1568x16] ----------------
    // 8 waves = 2 M-tiles x 4 K-chunks; ds_add_f32 reduction in LDS.
    {
        const int mt = wave & 1;
        const int kc = wave >> 1;
        const int o  = mt * 16 + (lane & 15);
        const int n  = lane & 15;
        v8f acc = {};
        for (int kk = kc; kk < 49; kk += 4) {    // one 7x7 tap per K-step
            int wi = kk / 7, wj = kk % 7;
            _Float16 gh = (_Float16)(1.f + s_att[n * 49 + kk]); // gate, once/frag
            const _Float16* bp = s_slab + ((wi * SLABW + n + wj) * CH + lg8);
            v16h bf = cat8(*(const v8h*)bp, *(const v8h*)(bp + 16)) * bcast16(gh);
            const _Float16* ap = s_w3t + ((o * 49 + kk) * CH + lg8);
            v16h af = cat8(*(const v8h*)ap, *(const v8h*)(ap + 16));
            acc = wmma_f16(af, bf, acc);
        }
        #pragma unroll
        for (int r = 0; r < 8; ++r)
            atomicAdd(&s_out[(mt * 16 + r + lg8) * 16 + n], acc[r]);
    }
    __syncthreads();

    // ---- Writeback: output[b,o,h,w0+n] = s_out[o][n] + b3[o] -------------
    for (int i = tid; i < 512; i += 256) {
        int o = i >> 4, n = i & 15;
        out[(((size_t)bimg * CH + o) * IMG + h) * IMG + w0 + n] = s_out[i] + b3[o];
    }
}

extern "C" void kernel_launch(void* const* d_in, const int* in_sizes, int n_in,
                              void* d_out, int out_size, void* d_ws, size_t ws_size,
                              hipStream_t stream) {
    const float* x  = (const float*)d_in[0];
    const float* w1 = (const float*)d_in[1];
    const float* b1 = (const float*)d_in[2];
    const float* w2 = (const float*)d_in[3];
    const float* b2 = (const float*)d_in[4];
    const float* w3 = (const float*)d_in[5];
    const float* b3 = (const float*)d_in[6];
    float* out     = (float*)d_out;
    float* att_out = out + (size_t)2 * 32 * 128 * 128;   // outputs concat'd flat

    dim3 grid(2 * 128 * (128 / TILE));   // 2048 workgroups, 16 pixels each
    dim3 block(256);                     // 8 wave32 waves
    hipLaunchKernelGGL(local_attn_kernel, grid, block, SMEM_BYTES, stream,
                       x, w1, b1, w2, b2, w3, b3, out, att_out);
}
